// SegNet_56959856279684
// MI455X (gfx1250) — matrix operand order repaired and checked by reading, not compiled, not run on hardware
//
#include <hip/hip_runtime.h>
#include <hip/hip_bf16.h>
#include <stdint.h>
#include <stddef.h>

// ---------------------------------------------------------------------------
// PointNet++ segmentation forward for MI455X (gfx1250, wave32, WMMA f16).
// Weights pre-packed to B-fragment layout (f16); LDS activations kept in
// A-fragment layout so every WMMA operand is a single contiguous v16h load.
// ---------------------------------------------------------------------------

#define BN_INV_C 0.9999950000374997f   // 1/sqrt(1+1e-5), eval-mode BatchNorm

#define BB   8
#define PPTS 4096
#define MM1  819
#define MM2  204
#define KNB  64
#define NCLS 13

typedef __attribute__((ext_vector_type(16))) _Float16 v16h;
typedef __attribute__((ext_vector_type(8)))  float    v8f;

union Acc8 { v8f v; float f[8]; };

// A-fragment address for element (row, k) of a 64-row LDS activation tile.
// nkc = number of 32-wide K chunks. Lane = hi*16 + (row&15); within-lane
// element e packs K as [0..7 | 16..23] (+8 per K-half).
__device__ __forceinline__ int a_frag_addr(int row, int k, int nkc)
{
    int rt = row >> 4, nn = row & 15;
    int kc = k >> 5,  kl = k & 31;
    int hi = (kl >> 3) & 1;
    int e  = (kl & 7) | ((kl >> 4) << 3);
    return (((rt * nkc + kc) * 32) + (hi * 16 + nn)) * 16 + e;
}

// ===========================================================================
// Weight packing: f32 [K x N] row-major -> f16 B-fragment layout,
// zero-padded to Kpad (mult 32) x Npad (mult 16).
// B layout: lane = ((k>>3)&1) * 16 + (n&15); element = (k&7) | (((k&31)>>4)<<3).
// ===========================================================================
__global__ void pack_w_kernel(const float* __restrict__ W, int K, int N,
                              int Kpad, int Npad, _Float16* __restrict__ out)
{
    int e = blockIdx.x * blockDim.x + threadIdx.x;
    int total = Kpad * Npad;
    if (e >= total) return;
    int k = e / Npad, n = e - k * Npad;
    int kc = k >> 5, kl = k & 31, ct = n >> 4, nn = n & 15;
    int addr = (((kc * (Npad >> 4) + ct) * 32) + (((kl >> 3) & 1) * 16 + nn)) * 16 + ((kl & 7) | ((kl >> 4) << 3));
    out[addr] = (k < K && n < N) ? (_Float16)W[(size_t)k * N + n] : (_Float16)0.0f;
}

// ===========================================================================
// FPS: one block per cloud; mind[] lives in LDS, serial argmax chain.
// ===========================================================================
__global__ void fps_kernel(const float* __restrict__ pos, int Pn, int m,
                           int* __restrict__ idx_out)
{
    extern __shared__ float smem[];
    float* mind = smem;                 // [Pn]
    float* rv   = smem + Pn;            // [256]
    int*   ri   = (int*)(rv + 256);     // [256]

    const int b   = blockIdx.x;
    const int tid = threadIdx.x;
    const float* pb = pos + (size_t)b * Pn * 3;

    for (int p = tid; p < Pn; p += 256) mind[p] = 1e10f;
    if (tid == 0) idx_out[b * m] = 0;
    __syncthreads();

    int last = 0;
    for (int i = 1; i < m; ++i) {
        const float lx = pb[last * 3 + 0];
        const float ly = pb[last * 3 + 1];
        const float lz = pb[last * 3 + 2];
        float bv = -1.0f; int bi = 0x7fffffff;
        for (int p = tid; p < Pn; p += 256) {
            float dx = pb[p * 3 + 0] - lx;
            float dy = pb[p * 3 + 1] - ly;
            float dz = pb[p * 3 + 2] - lz;
            float d  = dx * dx + dy * dy + dz * dz;
            float mn = fminf(mind[p], d);
            mind[p] = mn;
            if (mn > bv) { bv = mn; bi = p; }
        }
        rv[tid] = bv; ri[tid] = bi;
        __syncthreads();
        for (int s = 128; s > 0; s >>= 1) {
            if (tid < s) {
                if (rv[tid + s] > rv[tid] ||
                    (rv[tid + s] == rv[tid] && ri[tid + s] < ri[tid])) {
                    rv[tid] = rv[tid + s]; ri[tid] = ri[tid + s];
                }
            }
            __syncthreads();
        }
        last = ri[0];
        if (tid == 0) idx_out[b * m + i] = last;
        __syncthreads();
    }
}

// ===========================================================================
__global__ void gather_centers_kernel(const float* __restrict__ pos_src,
                                      const int* __restrict__ idx,
                                      int Pn, int m, int Bn,
                                      float* __restrict__ cen)
{
    int e = blockIdx.x * blockDim.x + threadIdx.x;
    int total = Bn * m * 3;
    if (e >= total) return;
    int d = e % 3;
    int r = e / 3;
    int b = r / m;
    cen[e] = pos_src[((size_t)b * Pn + idx[r]) * 3 + d];
}

// ===========================================================================
// K=64 nearest neighbors per center (iterative argmin over LDS d2 array).
// ===========================================================================
__global__ void neighbors_kernel(const float* __restrict__ pos_src,
                                 const float* __restrict__ cen,
                                 int Pn, int m, float r2,
                                 int* __restrict__ nidx,
                                 int* __restrict__ nmask)
{
    extern __shared__ float smem[];
    float* d2 = smem;                  // [Pn]
    float* rv = smem + Pn;             // [256]
    int*   ri = (int*)(rv + 256);      // [256]

    const int c   = blockIdx.x;        // global center id (b*m + i)
    const int b   = c / m;
    const int tid = threadIdx.x;
    const float* pb = pos_src + (size_t)b * Pn * 3;

    const float cx = cen[c * 3 + 0];
    const float cy = cen[c * 3 + 1];
    const float cz = cen[c * 3 + 2];

    for (int p = tid; p < Pn; p += 256) {
        float dx = pb[p * 3 + 0] - cx;
        float dy = pb[p * 3 + 1] - cy;
        float dz = pb[p * 3 + 2] - cz;
        d2[p] = dx * dx + dy * dy + dz * dz;
    }
    __syncthreads();

    for (int k = 0; k < KNB; ++k) {
        float bv = 3.0e38f; int bi = 0x7fffffff;
        for (int p = tid; p < Pn; p += 256) {
            float d = d2[p];
            if (d < bv) { bv = d; bi = p; }
        }
        rv[tid] = bv; ri[tid] = bi;
        __syncthreads();
        for (int s = 128; s > 0; s >>= 1) {
            if (tid < s) {
                if (rv[tid + s] < rv[tid] ||
                    (rv[tid + s] == rv[tid] && ri[tid + s] < ri[tid])) {
                    rv[tid] = rv[tid + s]; ri[tid] = ri[tid + s];
                }
            }
            __syncthreads();
        }
        if (tid == 0) {
            int   sel = ri[0];
            float dv  = rv[0];
            nidx[(size_t)c * KNB + k]  = sel;
            nmask[(size_t)c * KNB + k] = (dv <= r2) ? 1 : 0;
            d2[sel] = 6.9e38f;         // exclude from later rounds
        }
        __syncthreads();
    }
}

// ===========================================================================
// One MLP layer on a 64-row LDS activation tile (A-fragment layout, f16)
// with pre-packed f16 weights (B-fragment layout). 8 waves per block.
// Inner loop: 1 LDS v16h load + 1 global v16h load + 1 WMMA per K-chunk.
// ===========================================================================
__device__ __forceinline__ void lds_layer(const _Float16* Ain, int nkc,
                                          const _Float16* __restrict__ Wp,
                                          const float* __restrict__ bias,
                                          const float* __restrict__ gam,
                                          const float* __restrict__ bet,
                                          int N, _Float16* Aout, int tid)
{
    const int lane = tid & 31;
    const int wave = tid >> 5;
    const int nn   = lane & 15;
    const int hi   = lane >> 4;
    const int nt16 = N >> 4;
    const int nokc = N >> 5;           // K-chunks of the *output* tile
    const int ntiles = 4 * nt16;

    for (int t = wave; t < ntiles; t += 8) {
        const int rt = t & 3;
        const int ct = t >> 2;
        Acc8 acc;
#pragma unroll
        for (int i = 0; i < 8; ++i) acc.f[i] = 0.0f;

        const _Float16* ap = Ain + (((size_t)rt * nkc) * 32 + lane) * 16;
        const _Float16* bp = Wp  + ((size_t)ct * 32 + lane) * 16;
        for (int kc = 0; kc < nkc; ++kc) {
            v16h a = *(const v16h*)(ap + (size_t)kc * 512);
            v16h b = *(const v16h*)(bp + (size_t)kc * nt16 * 512);
            acc.v = __builtin_amdgcn_wmma_f32_16x16x32_f16(
                false, a, false, b, (short)0, acc.v, false, false);
        }
#pragma unroll
        for (int i = 0; i < 8; ++i) {
            int row = rt * 16 + hi * 8 + i;          // C/D layout
            int col = ct * 16 + nn;
            float v = acc.f[i] + bias[col];
            v = fmaxf(v, 0.0f);                      // ReLU
            v = v * (gam[col] * BN_INV_C) + bet[col];// eval BN
            Aout[a_frag_addr(row, col, nokc)] = (_Float16)v;
        }
    }
}

// ===========================================================================
// Fused PointConv SA module: one block (256 thr / 8 waves) per center.
// Gather [64 x (Fin+3)] -> 3 WMMA MLP layers (LDS-resident) -> masked max.
// ===========================================================================
__global__ void __launch_bounds__(256)
sa_pointconv_kernel(const float* __restrict__ x_src,
                    const float* __restrict__ pos_src,
                    const float* __restrict__ cen,
                    const int* __restrict__ nidx,
                    const int* __restrict__ nmask,
                    int Pn, int m, int Fin,
                    const _Float16* W1, const float* b1, const float* g1, const float* t1, int C1,
                    const _Float16* W2, const float* b2, const float* g2, const float* t2, int C2,
                    const _Float16* W3, const float* b3, const float* g3, const float* t3, int C3,
                    float* __restrict__ x_out)
{
    __shared__ alignas(32) _Float16 bufA[64 * 160];  // input / layer-2 out
    __shared__ alignas(32) _Float16 bufB[64 * 256];  // layer-1 out / layer-3 out
    __shared__ int smask[64];

    const int c   = blockIdx.x;          // global center id
    const int b   = c / m;
    const int tid = threadIdx.x;
    const int K0   = Fin + 3;
    const int Kpad = (K0 + 31) & ~31;
    const int nkc0 = Kpad >> 5;

    const float cx = cen[c * 3 + 0];
    const float cy = cen[c * 3 + 1];
    const float cz = cen[c * 3 + 2];

    if (tid < 64) smask[tid] = nmask[(size_t)c * KNB + tid];

    // gather neighbor features + relative positions into A-fragment layout
    for (int e = tid; e < 64 * Kpad; e += 256) {
        int j   = e / Kpad;
        int col = e - j * Kpad;
        int pidx = nidx[(size_t)c * KNB + j];
        float v = 0.0f;
        if (col < Fin) {
            v = x_src[((size_t)b * Pn + pidx) * Fin + col];
        } else if (col < K0) {
            int d = col - Fin;
            float pv = pos_src[((size_t)b * Pn + pidx) * 3 + d];
            v = pv - ((d == 0) ? cx : (d == 1) ? cy : cz);
        }
        bufA[a_frag_addr(j, col, nkc0)] = (_Float16)v;
    }
    __syncthreads();

    lds_layer(bufA, nkc0,    W1, b1, g1, t1, C1, bufB, tid);
    __syncthreads();
    lds_layer(bufB, C1 >> 5, W2, b2, g2, t2, C2, bufA, tid);
    __syncthreads();
    lds_layer(bufA, C2 >> 5, W3, b3, g3, t3, C3, bufB, tid);
    __syncthreads();

    // masked max over the 64 neighbors
    const int nkc3 = C3 >> 5;
    for (int ch = tid; ch < C3; ch += 256) {
        float mx = -1e10f; int any = 0;
        for (int j = 0; j < KNB; ++j) {
            if (smask[j]) {
                any = 1;
                float v = (float)bufB[a_frag_addr(j, ch, nkc3)];
                mx = fmaxf(mx, v);
            }
        }
        x_out[(size_t)c * C3 + ch] = any ? mx : 0.0f;
    }
}

// ===========================================================================
// Generic GEMM + bias (+optional ReLU, +optional eval-BN).
//   A[M x K] f32; Wp packed f16 (Kpad x Npad, B-fragment layout);
//   Out[M x N] f32. block = 128 (4 waves); grid = (ceil(M/16), ceil(Npad/64)).
// ===========================================================================
__global__ void __launch_bounds__(128)
gemm_kernel(const float* __restrict__ A, int M, int K,
            const _Float16* __restrict__ Wp,
            const float* __restrict__ bias,
            const float* __restrict__ gam, const float* __restrict__ bet,
            float* __restrict__ Out, int N, int Npad, int do_relu, int do_bn)
{
    __shared__ alignas(32) _Float16 tileA[512];     // 16x32, A-fragment layout

    const int tid  = threadIdx.x;
    const int lane = tid & 31;
    const int wave = tid >> 5;
    const int nn   = lane & 15;
    const int hi   = lane >> 4;
    const int r0   = blockIdx.x * 16;
    const int c0   = blockIdx.y * 64 + wave * 16;
    const int ct   = c0 >> 4;
    const int nt16 = Npad >> 4;

    Acc8 acc;
#pragma unroll
    for (int i = 0; i < 8; ++i) acc.f[i] = 0.0f;

    int kc = 0;
    for (int kb = 0; kb < K; kb += 32, ++kc) {
        // stage the 16x32 A tile directly in fragment order (f32 -> f16)
        if (r0 + 16 <= M && kb + 32 <= K) {
            for (int idx = tid; idx < 512; idx += 128) {
                int lt = idx >> 4, e = idx & 15;
                int tnn = lt & 15, thi = lt >> 4;
                int kl = ((e < 8) ? e : (e + 8)) + thi * 8;
                tileA[idx] = (_Float16)A[(size_t)(r0 + tnn) * K + kb + kl];
            }
        } else {
            for (int idx = tid; idx < 512; idx += 128) {
                int lt = idx >> 4, e = idx & 15;
                int tnn = lt & 15, thi = lt >> 4;
                int kl = ((e < 8) ? e : (e + 8)) + thi * 8;
                int row = r0 + tnn, k = kb + kl;
                tileA[idx] = (row < M && k < K) ? (_Float16)A[(size_t)row * K + k]
                                                : (_Float16)0.0f;
            }
        }
        __syncthreads();

        if (c0 < Npad) {
            if (lane == 0 && kb + 32 < K)
                __builtin_prefetch(Wp + (((size_t)(kc + 1) * nt16 + ct) * 32) * 16, 0, 0);
            v16h a = *(const v16h*)(tileA + lane * 16);
            v16h b = *(const v16h*)(Wp + (((size_t)kc * nt16 + ct) * 32 + lane) * 16);
            acc.v = __builtin_amdgcn_wmma_f32_16x16x32_f16(
                false, a, false, b, (short)0, acc.v, false, false);
        }
        __syncthreads();
    }

    if (c0 < N) {
#pragma unroll
        for (int i = 0; i < 8; ++i) {
            int row = r0 + hi * 8 + i;
            int col = c0 + nn;
            if (row < M && col < N) {
                float v = acc.f[i] + bias[col];
                if (do_relu) v = fmaxf(v, 0.0f);
                if (do_bn)   v = v * (gam[col] * BN_INV_C) + bet[col];
                Out[(size_t)row * N + col] = v;
            }
        }
    }
}

// ===========================================================================
// Misc elementwise kernels
// ===========================================================================
__global__ void concat_xpos_kernel(const float* __restrict__ x,
                                   const float* __restrict__ pos,
                                   int rows, int F, float* __restrict__ out)
{
    int e = blockIdx.x * blockDim.x + threadIdx.x;
    int Ft = F + 3;
    if (e >= rows * Ft) return;
    int r = e / Ft, c = e - r * Ft;
    out[e] = (c < F) ? x[(size_t)r * F + c] : pos[(size_t)r * 3 + (c - F)];
}

__global__ void maxpool_rows_kernel(const float* __restrict__ h, int per, int F,
                                    int Bn, float* __restrict__ out)
{
    int e = blockIdx.x * blockDim.x + threadIdx.x;
    if (e >= Bn * F) return;
    int b = e / F, c = e - b * F;
    float mx = -3.0e38f;
    for (int j = 0; j < per; ++j)
        mx = fmaxf(mx, h[((size_t)b * per + j) * F + c]);
    out[e] = mx;
}

__global__ void concat_bcast_kernel(const float* __restrict__ g,
                                    const float* __restrict__ x2,
                                    int rows, int per, int Fg, int Fx,
                                    float* __restrict__ out)
{
    int e = blockIdx.x * blockDim.x + threadIdx.x;
    int Ft = Fg + Fx;
    if (e >= rows * Ft) return;
    int r = e / Ft, c = e - r * Ft;
    int b = r / per;
    out[e] = (c < Fg) ? g[(size_t)b * Fg + c] : x2[(size_t)r * Fx + (c - Fg)];
}

__global__ void knn3_kernel(const float* __restrict__ pt,
                            const float* __restrict__ ps,
                            int T, int S, int Bn,
                            int* __restrict__ oidx, float* __restrict__ ow)
{
    int t = blockIdx.x * blockDim.x + threadIdx.x;
    if (t >= Bn * T) return;
    int b = t / T;
    float tx = pt[(size_t)t * 3 + 0];
    float ty = pt[(size_t)t * 3 + 1];
    float tz = pt[(size_t)t * 3 + 2];
    float d0 = 3e38f, d1 = 3e38f, d2v = 3e38f;
    int   i0 = 0,     i1 = 0,     i2 = 0;
    const float* pb = ps + (size_t)b * S * 3;
    for (int s = 0; s < S; ++s) {
        float dx = pb[s * 3 + 0] - tx;
        float dy = pb[s * 3 + 1] - ty;
        float dz = pb[s * 3 + 2] - tz;
        float d  = fmaxf(dx * dx + dy * dy + dz * dz, 0.0f);
        if (d < d0)      { d2v = d1; i2 = i1; d1 = d0; i1 = i0; d0 = d; i0 = s; }
        else if (d < d1) { d2v = d1; i2 = i1; d1 = d;  i1 = s; }
        else if (d < d2v){ d2v = d;  i2 = s; }
    }
    float w0 = 1.0f / fmaxf(d0, 1e-16f);
    float w1 = 1.0f / fmaxf(d1, 1e-16f);
    float w2 = 1.0f / fmaxf(d2v, 1e-16f);
    float sw = w0 + w1 + w2;
    oidx[(size_t)t * 3 + 0] = i0; ow[(size_t)t * 3 + 0] = w0 / sw;
    oidx[(size_t)t * 3 + 1] = i1; ow[(size_t)t * 3 + 1] = w1 / sw;
    oidx[(size_t)t * 3 + 2] = i2; ow[(size_t)t * 3 + 2] = w2 / sw;
}

__global__ void interp_concat_kernel(const float* __restrict__ src,
                                     const int* __restrict__ idx3,
                                     const float* __restrict__ w3,
                                     const float* __restrict__ skip,
                                     int T, int S, int Fi, int Fs, int Bn,
                                     float* __restrict__ out)
{
    int F = Fi + Fs;
    int e = blockIdx.x * blockDim.x + threadIdx.x;
    if (e >= Bn * T * F) return;
    int r = e / F, c = e - r * F;
    if (c < Fi) {
        int b = r / T;
        float acc = 0.0f;
#pragma unroll
        for (int j = 0; j < 3; ++j) {
            int   id = idx3[(size_t)r * 3 + j];
            float w  = w3[(size_t)r * 3 + j];
            acc += w * src[((size_t)b * S + id) * Fi + c];
        }
        out[e] = acc;
    } else {
        out[e] = skip[(size_t)r * Fs + (c - Fi)];
    }
}

// ===========================================================================
// Host orchestration
// ===========================================================================
extern "C" void kernel_launch(void* const* d_in, const int* in_sizes, int n_in,
                              void* d_out, int out_size, void* d_ws, size_t ws_size,
                              hipStream_t stream)
{
    (void)in_sizes; (void)n_in; (void)out_size; (void)ws_size;

    // ---- input pointer map (setup_inputs dict order, nested structs flattened)
    const float* x0   = (const float*)d_in[0];      // [B*P, 1]
    const float* pos0 = (const float*)d_in[1];      // [B*P, 3]
    auto Lp = [&](int base, int layer, int comp) -> const float* {
        return (const float*)d_in[base + layer * 4 + comp];
    };
    const int SA1 = 3, SA2 = 15, SA3 = 27, FP3 = 39, FP2 = 47, FP1 = 55;
    const float* lin1W = (const float*)d_in[67]; const float* lin1b = (const float*)d_in[68];
    const float* lin2W = (const float*)d_in[69]; const float* lin2b = (const float*)d_in[70];
    const float* lin3W = (const float*)d_in[71]; const float* lin3b = (const float*)d_in[72];

    // ---- workspace bump allocator (256B aligned)
    char* wsb = (char*)d_ws;
    size_t off = 0;
    auto alloc = [&](size_t bytes) -> void* {
        off = (off + 255) & ~(size_t)255;
        void* p = wsb + off;
        off += bytes;
        return p;
    };
    auto ceil_div = [](int a, int b) { return (a + b - 1) / b; };

    // ---- pack all weight matrices to f16 B-fragment layout
    struct PW { _Float16* p; int Kpad; int Npad; };
    auto packw = [&](const float* W, int K, int N) -> PW {
        int Kpad = (K + 31) & ~31;
        int Npad = (N + 15) & ~15;
        _Float16* out = (_Float16*)alloc((size_t)Kpad * Npad * 2);
        int total = Kpad * Npad;
        pack_w_kernel<<<ceil_div(total, 256), 256, 0, stream>>>(W, K, N, Kpad, Npad, out);
        return PW{out, Kpad, Npad};
    };

    PW sa1w[3] = { packw(Lp(SA1,0,0),   4,  64), packw(Lp(SA1,1,0),  64,  64),
                   packw(Lp(SA1,2,0),  64, 128) };
    PW sa2w[3] = { packw(Lp(SA2,0,0), 131, 128), packw(Lp(SA2,1,0), 128, 128),
                   packw(Lp(SA2,2,0), 128, 256) };
    PW sa3w[3] = { packw(Lp(SA3,0,0), 259, 256), packw(Lp(SA3,1,0), 256, 512),
                   packw(Lp(SA3,2,0), 512, 1024) };
    PW fp3w[2] = { packw(Lp(FP3,0,0), 1280, 256), packw(Lp(FP3,1,0), 256, 256) };
    PW fp2w[2] = { packw(Lp(FP2,0,0),  384, 256), packw(Lp(FP2,1,0), 256, 128) };
    PW fp1w[3] = { packw(Lp(FP1,0,0),  129, 128), packw(Lp(FP1,1,0), 128, 128),
                   packw(Lp(FP1,2,0),  128, 128) };
    PW l1w = packw(lin1W, 128, 128);
    PW l2w = packw(lin2W, 128, 128);
    PW l3w = packw(lin3W, 128, NCLS);

    const int NC1 = BB * MM1;     // 6552 centers level-1
    const int NC2 = BB * MM2;     // 1632 centers level-2
    const int NP  = BB * PPTS;    // 32768 points

    int*   idx1      = (int*)  alloc((size_t)NC1 * 4);
    float* pos1      = (float*)alloc((size_t)NC1 * 3 * 4);
    int*   nbr1_idx  = (int*)  alloc((size_t)NC1 * KNB * 4);
    int*   nbr1_msk  = (int*)  alloc((size_t)NC1 * KNB * 4);
    float* x1        = (float*)alloc((size_t)NC1 * 128 * 4);
    int*   idx2      = (int*)  alloc((size_t)NC2 * 4);
    float* pos2      = (float*)alloc((size_t)NC2 * 3 * 4);
    int*   nbr2_idx  = (int*)  alloc((size_t)NC2 * KNB * 4);
    int*   nbr2_msk  = (int*)  alloc((size_t)NC2 * KNB * 4);
    float* x2        = (float*)alloc((size_t)NC2 * 256 * 4);
    float* sa3_in    = (float*)alloc((size_t)NC2 * 259 * 4);
    float* sa3_h1    = (float*)alloc((size_t)NC2 * 256 * 4);
    float* sa3_h2    = (float*)alloc((size_t)NC2 * 512 * 4);
    float* sa3_h3    = (float*)alloc((size_t)NC2 * 1024 * 4);
    float* gbuf      = (float*)alloc((size_t)BB * 1024 * 4);
    float* fp3_in    = (float*)alloc((size_t)NC2 * 1280 * 4);
    float* fp3_h1    = (float*)alloc((size_t)NC2 * 256 * 4);
    float* h3f       = (float*)alloc((size_t)NC2 * 256 * 4);
    int*   knn2_idx  = (int*)  alloc((size_t)NC1 * 3 * 4);
    float* knn2_w    = (float*)alloc((size_t)NC1 * 3 * 4);
    float* fp2_in    = (float*)alloc((size_t)NC1 * 384 * 4);
    float* fp2_h1    = (float*)alloc((size_t)NC1 * 256 * 4);
    float* h2f       = (float*)alloc((size_t)NC1 * 128 * 4);
    int*   knn1_idx  = (int*)  alloc((size_t)NP * 3 * 4);
    float* knn1_w    = (float*)alloc((size_t)NP * 3 * 4);
    float* fp1_in    = (float*)alloc((size_t)NP * 129 * 4);
    float* fp1_h1    = (float*)alloc((size_t)NP * 128 * 4);
    float* fp1_h2    = (float*)alloc((size_t)NP * 128 * 4);
    float* h1f       = (float*)alloc((size_t)NP * 128 * 4);
    float* lin1_o    = (float*)alloc((size_t)NP * 128 * 4);
    float* lin2_o    = (float*)alloc((size_t)NP * 128 * 4);

    auto gemm = [&](const float* A, int M, int K, const PW& w, const float* bias,
                    const float* ga, const float* be, float* O, int N,
                    int relu, int bn) {
        dim3 g(ceil_div(M, 16), ceil_div(w.Npad, 64));
        gemm_kernel<<<g, 128, 0, stream>>>(A, M, K, w.p, bias, ga, be, O,
                                           N, w.Npad, relu, bn);
    };

    // -------------------- SA1 --------------------
    fps_kernel<<<BB, 256, PPTS * 4 + 256 * 8, stream>>>(pos0, PPTS, MM1, idx1);
    gather_centers_kernel<<<ceil_div(NC1 * 3, 256), 256, 0, stream>>>(
        pos0, idx1, PPTS, MM1, BB, pos1);
    neighbors_kernel<<<NC1, 256, PPTS * 4 + 256 * 8, stream>>>(
        pos0, pos1, PPTS, MM1, 0.2f * 0.2f, nbr1_idx, nbr1_msk);
    sa_pointconv_kernel<<<NC1, 256, 0, stream>>>(
        x0, pos0, pos1, nbr1_idx, nbr1_msk, PPTS, MM1, 1,
        sa1w[0].p, Lp(SA1,0,1), Lp(SA1,0,2), Lp(SA1,0,3), 64,
        sa1w[1].p, Lp(SA1,1,1), Lp(SA1,1,2), Lp(SA1,1,3), 64,
        sa1w[2].p, Lp(SA1,2,1), Lp(SA1,2,2), Lp(SA1,2,3), 128,
        x1);

    // -------------------- SA2 --------------------
    fps_kernel<<<BB, 256, MM1 * 4 + 256 * 8, stream>>>(pos1, MM1, MM2, idx2);
    gather_centers_kernel<<<ceil_div(NC2 * 3, 256), 256, 0, stream>>>(
        pos1, idx2, MM1, MM2, BB, pos2);
    neighbors_kernel<<<NC2, 256, MM1 * 4 + 256 * 8, stream>>>(
        pos1, pos2, MM1, MM2, 0.4f * 0.4f, nbr2_idx, nbr2_msk);
    sa_pointconv_kernel<<<NC2, 256, 0, stream>>>(
        x1, pos1, pos2, nbr2_idx, nbr2_msk, MM1, MM2, 128,
        sa2w[0].p, Lp(SA2,0,1), Lp(SA2,0,2), Lp(SA2,0,3), 128,
        sa2w[1].p, Lp(SA2,1,1), Lp(SA2,1,2), Lp(SA2,1,3), 128,
        sa2w[2].p, Lp(SA2,2,1), Lp(SA2,2,2), Lp(SA2,2,3), 256,
        x2);

    // -------------------- SA3 (global) --------------------
    concat_xpos_kernel<<<ceil_div(NC2 * 259, 256), 256, 0, stream>>>(
        x2, pos2, NC2, 256, sa3_in);
    gemm(sa3_in, NC2, 259, sa3w[0], Lp(SA3,0,1), Lp(SA3,0,2), Lp(SA3,0,3),
         sa3_h1, 256, 1, 1);
    gemm(sa3_h1, NC2, 256, sa3w[1], Lp(SA3,1,1), Lp(SA3,1,2), Lp(SA3,1,3),
         sa3_h2, 512, 1, 1);
    gemm(sa3_h2, NC2, 512, sa3w[2], Lp(SA3,2,1), Lp(SA3,2,2), Lp(SA3,2,3),
         sa3_h3, 1024, 1, 1);
    maxpool_rows_kernel<<<ceil_div(BB * 1024, 256), 256, 0, stream>>>(
        sa3_h3, MM2, 1024, BB, gbuf);

    // -------------------- FP3 --------------------
    concat_bcast_kernel<<<ceil_div(NC2 * 1280, 256), 256, 0, stream>>>(
        gbuf, x2, NC2, MM2, 1024, 256, fp3_in);
    gemm(fp3_in, NC2, 1280, fp3w[0], Lp(FP3,0,1), Lp(FP3,0,2), Lp(FP3,0,3),
         fp3_h1, 256, 1, 1);
    gemm(fp3_h1, NC2, 256, fp3w[1], Lp(FP3,1,1), Lp(FP3,1,2), Lp(FP3,1,3),
         h3f, 256, 1, 1);

    // -------------------- FP2 --------------------
    knn3_kernel<<<ceil_div(NC1, 256), 256, 0, stream>>>(
        pos1, pos2, MM1, MM2, BB, knn2_idx, knn2_w);
    interp_concat_kernel<<<ceil_div(NC1 * 384, 256), 256, 0, stream>>>(
        h3f, knn2_idx, knn2_w, x1, MM1, MM2, 256, 128, BB, fp2_in);
    gemm(fp2_in, NC1, 384, fp2w[0], Lp(FP2,0,1), Lp(FP2,0,2), Lp(FP2,0,3),
         fp2_h1, 256, 1, 1);
    gemm(fp2_h1, NC1, 256, fp2w[1], Lp(FP2,1,1), Lp(FP2,1,2), Lp(FP2,1,3),
         h2f, 128, 1, 1);

    // -------------------- FP1 --------------------
    knn3_kernel<<<ceil_div(NP, 256), 256, 0, stream>>>(
        pos0, pos1, PPTS, MM1, BB, knn1_idx, knn1_w);
    interp_concat_kernel<<<ceil_div(NP * 129, 256), 256, 0, stream>>>(
        h2f, knn1_idx, knn1_w, x0, PPTS, MM1, 128, 1, BB, fp1_in);
    gemm(fp1_in, NP, 129, fp1w[0], Lp(FP1,0,1), Lp(FP1,0,2), Lp(FP1,0,3),
         fp1_h1, 128, 1, 1);
    gemm(fp1_h1, NP, 128, fp1w[1], Lp(FP1,1,1), Lp(FP1,1,2), Lp(FP1,1,3),
         fp1_h2, 128, 1, 1);
    gemm(fp1_h2, NP, 128, fp1w[2], Lp(FP1,2,1), Lp(FP1,2,2), Lp(FP1,2,3),
         h1f, 128, 1, 1);

    // -------------------- heads --------------------
    gemm(h1f,    NP, 128, l1w, lin1b, nullptr, nullptr, lin1_o, 128, 1, 0);
    gemm(lin1_o, NP, 128, l2w, lin2b, nullptr, nullptr, lin2_o, 128, 0, 0);
    gemm(lin2_o, NP, 128, l3w, lin3b, nullptr, nullptr, (float*)d_out, NCLS, 0, 0);
}
